// GraphSAGENet_56401510531240
// MI455X (gfx1250) — matrix operand order repaired and checked by reading, not compiled, not run on hardware
//
#include <hip/hip_runtime.h>
#include <hip/hip_bf16.h>
#include <stdint.h>

// ---------------------------------------------------------------------------
// GraphSAGE 2-layer forward on MI455X (gfx1250, wave32, WMMA).
//
// Layer math: out = lin_l(mean_agg) + lin_r(self) + b  ==  [agg|self] @ [Wl;Wr] + b
// GEMMs run on v_wmma_f32_16x16x32_f16 (f16 in, f32 accumulate).
// Scatter phase uses coalesced float4 row gathers + returnless f32 atomics
// (the 51MB accumulator lives in the 192MB L2, so atomics resolve in L2).
// ---------------------------------------------------------------------------

typedef __attribute__((ext_vector_type(16))) _Float16 v16h;
typedef __attribute__((ext_vector_type(8)))  _Float16 v8h;
typedef __attribute__((ext_vector_type(8)))  float    v8f;

#define AROW 280  // LDS row pitch in halves: 140 DWORDs -> stride 12 mod 64 banks,
                  // 16 rows hit 16 distinct 4-bank groups (conflict-free b128 reads)

// ---------------------------------------------------------------------------
// zero: float4 stores, tail-safe
// ---------------------------------------------------------------------------
__global__ void zero_kernel(float* __restrict__ p, long long n) {
  long long i = ((long long)blockIdx.x * blockDim.x + threadIdx.x) * 4;
  if (i + 3 < n) {
    *(float4*)(p + i) = make_float4(0.f, 0.f, 0.f, 0.f);
  } else {
    for (long long j = i; j < n; ++j) p[j] = 0.f;
  }
}

// ---------------------------------------------------------------------------
// pack [Wl;Wr] (f32, each [128,ncols]) into f16 WMMA-B-fragment layout:
//   Wpk[(kblock*ncols + n)*32 + kin]   with kblock = k>>5, kin = k&31 with bits 3 and 4 swapped
// Consumer lane (half=lane>>4, n=ntile+lane&15) then reads its 16 B-values
// (K{0..7,16..23} for half 0, K{8..15,24..31} for half 1, matching A) as ONE contiguous, 32B-aligned v16h load.
// ---------------------------------------------------------------------------
__global__ void pack_weights(const float* __restrict__ Wl,
                             const float* __restrict__ Wr,
                             _Float16* __restrict__ Wpk, int ncols) {
  int i = blockIdx.x * blockDim.x + threadIdx.x;
  int total = 256 * ncols;
  if (i >= total) return;
  int k = i / ncols;
  int n = i - k * ncols;
  float v = (k < 128) ? Wl[k * ncols + n] : Wr[(k - 128) * ncols + n];
  int kb  = k >> 5;
  int kin = (((k >> 3) & 1) << 4) | (k & 7) | (((k >> 4) & 1) << 3);
  Wpk[((size_t)(kb * ncols + n)) * 32 + kin] = (_Float16)v;
}

// ---------------------------------------------------------------------------
// degree: one thread per edge, atomic counting (no return -> returnless atomic)
// ---------------------------------------------------------------------------
__global__ void degree_kernel(const long long* __restrict__ dst,
                              float* __restrict__ deg, long long nE) {
  long long e = (long long)blockIdx.x * blockDim.x + threadIdx.x;
  if (e < nE) atomicAdd(&deg[dst[e]], 1.0f);
}

__global__ void recip_deg(float* __restrict__ deg, int n) {
  int i = blockIdx.x * blockDim.x + threadIdx.x;
  if (i < n) deg[i] = 1.0f / fmaxf(deg[i], 1.0f);
}

// ---------------------------------------------------------------------------
// scatter-add: one wave per edge. 32 lanes x float4 = one full 512B source row
// per wave (perfectly coalesced b128 loads), then 4 returnless f32 atomics/lane
// into the L2-resident accumulator.
// ---------------------------------------------------------------------------
__global__ void scatter_add_rows(const float* __restrict__ feat,
                                 const long long* __restrict__ src,
                                 const long long* __restrict__ dst,
                                 float* __restrict__ agg, long long nE) {
  const int lane = threadIdx.x & 31;
  long long e = (long long)blockIdx.x * (blockDim.x >> 5) + (threadIdx.x >> 5);
  if (e >= nE) return;
  long long s = src[e];
  long long d = dst[e];
  float4 v = *(const float4*)(feat + s * 128 + lane * 4);
  float* dp = agg + d * 128 + lane * 4;
  atomicAdd(dp + 0, v.x);
  atomicAdd(dp + 1, v.y);
  atomicAdd(dp + 2, v.z);
  atomicAdd(dp + 3, v.w);
}

// ---------------------------------------------------------------------------
// Fused SAGE GEMM:  out[m, :] = relu?( [agg[m]*invd[m] | self[m]] @ Wpk + b )
//   - block = 16 rows (M) x full ncols (N);  one wave per 16x16 N-tile
//   - K = 256, 8 x v_wmma_f32_16x16x32_f16 per wave
//   - A tile staged in LDS pre-swizzled to the ISA A-fragment layout
//     (16-bit A 16x32: lane<16 holds K{0..7,16..23}, lane>=16 holds K{8..15,24..31})
//     so each lane's 16 A-halves are contiguous -> two 16B ds loads.
// ---------------------------------------------------------------------------
__global__ void sage_gemm_wmma(const float* __restrict__ agg,
                               const float* __restrict__ selfm,
                               const float* __restrict__ invd,
                               const _Float16* __restrict__ Wpk,
                               const float* __restrict__ bias,
                               float* __restrict__ out,
                               int n_nodes, int ncols, int do_relu) {
  __shared__ _Float16 As[16 * AROW];

  const int tid = threadIdx.x;
  const int nth = blockDim.x;
  const int m0  = blockIdx.x * 16;

  // Stage A tile (16 rows x K=256) as f16 in fragment layout.
  // Fragment position of K=c: kb=c>>5, half=(c>>3)&1, idx=(c&7)|(((c>>4)&1)<<3)
  // -> for c4 % 4 == 0 the four elements land contiguously (one 8B LDS store).
  for (int i = tid; i < 16 * 64; i += nth) {
    int r   = i >> 6;
    int c4  = (i & 63) << 2;
    int node = m0 + r;
    float4 v = make_float4(0.f, 0.f, 0.f, 0.f);
    if (node < n_nodes) {
      if (c4 < 128) {                       // mean-aggregated neighbor part
        v = *(const float4*)(agg + (size_t)node * 128 + c4);
        float s = invd[node];
        v.x *= s; v.y *= s; v.z *= s; v.w *= s;
      } else {                              // self (root) part
        v = *(const float4*)(selfm + (size_t)node * 128 + (c4 - 128));
      }
    }
    int kb   = c4 >> 5;
    int hf   = (c4 >> 3) & 1;
    int idx  = (c4 & 7) | (((c4 >> 4) & 1) << 3);
    int ppos = r * AROW + kb * 32 + hf * 16 + idx;
    As[ppos + 0] = (_Float16)v.x;
    As[ppos + 1] = (_Float16)v.y;
    As[ppos + 2] = (_Float16)v.z;
    As[ppos + 3] = (_Float16)v.w;
  }
  __syncthreads();

  const int wave  = tid >> 5;
  const int lane  = tid & 31;
  const int hh    = lane >> 4;   // which half of the wave
  const int lrow  = lane & 15;
  const int ntile = wave * 16;

  v8f acc = {};
#pragma unroll
  for (int kb = 0; kb < 8; ++kb) {
    // A fragment: 16 contiguous halves (16B-aligned) -> two v8h LDS loads
    const v8h* pa = (const v8h*)(As + lrow * AROW + kb * 32 + hh * 16);
    v8h a0 = pa[0];
    v8h a1 = pa[1];
    v16h av = __builtin_shufflevector(a0, a1,
        0, 1, 2, 3, 4, 5, 6, 7, 8, 9, 10, 11, 12, 13, 14, 15);
    // B fragment: 16 contiguous halves, 32B-aligned global load
    v16h bv = *(const v16h*)(Wpk + ((size_t)(kb * ncols + ntile + lrow)) * 32 + hh * 16);
    acc = __builtin_amdgcn_wmma_f32_16x16x32_f16(
        /*neg_a=*/false, av, /*neg_b=*/false, bv,
        /*c_mod=*/(short)0, acc, /*reuse_a=*/false, /*reuse_b=*/false);
  }

  // C/D layout: VGPR r holds M = r + hh*8, N = lrow (per 16x16 f32 table)
  union { v8f v; float f[8]; } u;
  u.v = acc;
  int col = ntile + lrow;
  float bb = bias[col];
#pragma unroll
  for (int r = 0; r < 8; ++r) {
    int row = m0 + hh * 8 + r;
    float val = u.f[r] + bb;
    if (do_relu) val = fmaxf(val, 0.f);
    if (row < n_nodes) out[(size_t)row * ncols + col] = val;
  }
}

// ---------------------------------------------------------------------------
// launch
// ---------------------------------------------------------------------------
extern "C" void kernel_launch(void* const* d_in, const int* in_sizes, int n_in,
                              void* d_out, int out_size, void* d_ws, size_t ws_size,
                              hipStream_t stream) {
  (void)n_in; (void)out_size; (void)ws_size;

  const float*     x    = (const float*)d_in[0];
  const long long* ei   = (const long long*)d_in[1];
  const float*     Wl1  = (const float*)d_in[2];
  const float*     Wr1  = (const float*)d_in[3];
  const float*     b1   = (const float*)d_in[4];
  const float*     Wl2  = (const float*)d_in[5];
  const float*     Wr2  = (const float*)d_in[6];
  const float*     b2   = (const float*)d_in[7];
  float*           out  = (float*)d_out;

  const int       IN_C = 128, HID_C = 128, OUT_C = 64;
  const int       N = in_sizes[0] / IN_C;
  const long long E = in_sizes[1] / 2;
  const long long* srci = ei;
  const long long* dsti = ei + E;

  // workspace carve-out (256B aligned)
  uintptr_t p = (uintptr_t)d_ws;
  auto up = [](uintptr_t v) { return (v + 255) & ~(uintptr_t)255; };
  float* agg = (float*)p;      p = up(p + (size_t)N * HID_C * sizeof(float));
  float* h   = (float*)p;      p = up(p + (size_t)N * HID_C * sizeof(float));
  float* dg  = (float*)p;      p = up(p + (size_t)N * sizeof(float));
  _Float16* wpk1 = (_Float16*)p; p = up(p + (size_t)256 * HID_C * sizeof(_Float16));
  _Float16* wpk2 = (_Float16*)p; p = up(p + (size_t)256 * OUT_C * sizeof(_Float16));

  const int TPB = 256;
  const long long nAgg = (long long)N * 128;
  const unsigned gZeroAgg = (unsigned)((nAgg / 4 + TPB - 1) / TPB);
  const unsigned gZeroDeg = (unsigned)(((N + 3) / 4 + TPB - 1) / TPB);
  const unsigned gEdge1   = (unsigned)((E + TPB - 1) / TPB);
  const unsigned gScatter = (unsigned)((E + 7) / 8);       // 8 edges/block (wave per edge)
  const unsigned gNodes   = (unsigned)((N + TPB - 1) / TPB);
  const unsigned gGemm    = (unsigned)((N + 15) / 16);

  // ---- setup: zero accumulators, pack weights to f16 fragment layout ----
  zero_kernel<<<gZeroAgg, TPB, 0, stream>>>(agg, nAgg);
  zero_kernel<<<gZeroDeg, TPB, 0, stream>>>(dg, (long long)N);
  pack_weights<<<(256 * HID_C + TPB - 1) / TPB, TPB, 0, stream>>>(Wl1, Wr1, wpk1, HID_C);
  pack_weights<<<(256 * OUT_C + TPB - 1) / TPB, TPB, 0, stream>>>(Wl2, Wr2, wpk2, OUT_C);

  // ---- layer 1 ----
  degree_kernel<<<gEdge1, TPB, 0, stream>>>(dsti, dg, E);
  scatter_add_rows<<<gScatter, TPB, 0, stream>>>(x, srci, dsti, agg, E);
  recip_deg<<<gNodes, TPB, 0, stream>>>(dg, N);
  sage_gemm_wmma<<<gGemm, (HID_C / 16) * 32, 0, stream>>>(
      agg, x, dg, wpk1, b1, h, N, HID_C, /*relu=*/1);

  // ---- layer 2 ----
  zero_kernel<<<gZeroAgg, TPB, 0, stream>>>(agg, nAgg);
  scatter_add_rows<<<gScatter, TPB, 0, stream>>>(h, srci, dsti, agg, E);
  sage_gemm_wmma<<<gGemm, (OUT_C / 16) * 32, 0, stream>>>(
      agg, h, dg, wpk2, b2, out, N, OUT_C, /*relu=*/0);
}
